// MultiheadLinearAttention_69080253989534
// MI455X (gfx1250) — compile-verified, matrix operand order checked by reading, not run on hardware
//
#include <hip/hip_runtime.h>
#include <hip/hip_bf16.h>
#include <math.h>

// ---------------------------------------------------------------------------
// MultiheadLinearAttention (RFF linear attention) for gfx1250 / MI455X.
// bf16 WMMA (v_wmma_f32_16x16x32_bf16) for all GEMM-shaped work; split
// parallel f32 reduction for Kv/ksum; z normalizer fused into the attention
// WMMA as an extra accumulator column + ds_bpermute broadcast.
// ---------------------------------------------------------------------------

typedef __attribute__((ext_vector_type(16))) __bf16 v16bf;
typedef __attribute__((ext_vector_type(8)))  __bf16 v8bf;
typedef __attribute__((ext_vector_type(4)))  __bf16 v4bf;
typedef __attribute__((ext_vector_type(8)))  float  v8f;

#define NDIM   1024
#define NHEADS 16
#define NDK    64
#define NRFF   128
#define NFEAT  256      // 2*NRFF
#define NSEQ   8192
#define NROWS  16384    // B*N = 2*8192
#define NBH    32       // B*HEADS
#define KVCHUNK 32      // sequence chunks per (b,h) for Kv partial sums
#define KVSPAN  (NSEQ / KVCHUNK)   // 256

// ---- fragment helpers ------------------------------------------------------
// A fragment (16x32 bf16, MxK): lane L<16 -> row row0+L, K chunks {k..k+7} and
// {k+16..k+23} with k = k0 + 8*(L>=16). B fragment mirrors it (lane -> column).
__device__ __forceinline__ v16bf load_frag(const __bf16* __restrict__ base,
                                           int ld, int row0, int k0, int lane) {
  const __bf16* p = base + (size_t)(row0 + (lane & 15)) * ld + k0 + ((lane >> 4) << 3);
  v8bf lo = *(const v8bf*)p;
  v8bf hi = *(const v8bf*)(p + 16);
  v16bf f;
#pragma unroll
  for (int i = 0; i < 8; ++i) { f[i] = lo[i]; f[8 + i] = hi[i]; }
  return f;
}

__device__ __forceinline__ v8f wmma_bf16(v16bf a, v16bf b, v8f c) {
  return __builtin_amdgcn_wmma_f32_16x16x32_bf16(false, a, false, b,
                                                 (short)0, c, false, false);
}

// ---- conversion kernels (vectorized: float4 in, 4x bf16 out) ---------------
__global__ void cvt_f32_bf16(const float* __restrict__ in, __bf16* __restrict__ out, int n4) {
  for (int i = blockIdx.x * blockDim.x + threadIdx.x; i < n4; i += gridDim.x * blockDim.x) {
    float4 fv = ((const float4*)in)[i];
    v4bf o;
    o[0] = (__bf16)fv.x; o[1] = (__bf16)fv.y; o[2] = (__bf16)fv.z; o[3] = (__bf16)fv.w;
    ((v4bf*)out)[i] = o;
  }
}

// Wr [H][DK][RFF] f32  ->  WrT [H][RFF][DK] bf16 (so K=d is contiguous)
__global__ void wr_transpose(const float* __restrict__ Wr, __bf16* __restrict__ WrT) {
  int i = blockIdx.x * blockDim.x + threadIdx.x;      // 16*128*64 = 131072
  if (i >= NHEADS * NRFF * NDK) return;
  int h = i >> 13;
  int rem = i & 8191;
  int r = rem >> 6;
  int d = rem & 63;
  WrT[i] = (__bf16)Wr[((size_t)h * NDK + d) * NRFF + r];
}

// ---- big GEMM: out[M=16384, N=1024] = A[16384,1024] @ W[1024,1024]^T -------
// one 32x64 tile per wave (2 A-frags x 4 B-frags = 8 WMMAs per k-step);
// consecutive waves in a block share the same B strip -> WGP$ hits.
__global__ void gemm_xwT(const __bf16* __restrict__ A, const __bf16* __restrict__ W,
                         __bf16* __restrict__ outB, float* __restrict__ outF,
                         const float* __restrict__ bias) {
  const int lane = threadIdx.x & 31;
  const int wave = blockIdx.x * (blockDim.x >> 5) + (threadIdx.x >> 5);
  const int nt = wave >> 9;         // 0..15   (same for 8 waves of a block)
  const int mt = wave & 511;        // 0..511
  const int row0 = mt * 32;
  const int col0 = nt * 64;

  v8f c[2][4];
#pragma unroll
  for (int i = 0; i < 2; ++i)
#pragma unroll
    for (int n = 0; n < 4; ++n)
#pragma unroll
      for (int r = 0; r < 8; ++r) c[i][n][r] = 0.f;

  for (int k0 = 0; k0 < NDIM; k0 += 32) {
    v16bf a0 = load_frag(A, NDIM, row0, k0, lane);
    v16bf a1 = load_frag(A, NDIM, row0 + 16, k0, lane);
#pragma unroll
    for (int n = 0; n < 4; ++n) {
      v16bf b = load_frag(W, NDIM, col0 + n * 16, k0, lane);
      c[0][n] = wmma_bf16(a0, b, c[0][n]);
      c[1][n] = wmma_bf16(a1, b, c[1][n]);
    }
  }

  const int mhi = 8 * (lane >> 4);
  const int ncol = lane & 15;
#pragma unroll
  for (int i = 0; i < 2; ++i)
#pragma unroll
    for (int n = 0; n < 4; ++n)
#pragma unroll
      for (int r = 0; r < 8; ++r) {
        size_t row = (size_t)(row0 + i * 16 + mhi + r);
        size_t col = (size_t)(col0 + n * 16 + ncol);
        if (outF) outF[row * NDIM + col] = c[i][n][r] + bias[col];
        else      outB[row * NDIM + col] = (__bf16)c[i][n][r];
      }
}

// ---- RFF features: feat[row][h*256 + {r, 128+r}] = cos/sin(q@WrT + br)/sqrt(R)
// per wave: 16 rows x 64 r's; consecutive waves share (h, rt) -> shared B.
__global__ void rff_kernel(const __bf16* __restrict__ Q, const __bf16* __restrict__ WrT,
                           const float* __restrict__ br, __bf16* __restrict__ feat) {
  const int lane = threadIdx.x & 31;
  const int wave = blockIdx.x * (blockDim.x >> 5) + (threadIdx.x >> 5);
  const int ht = wave >> 10;        // 0..31 (same for the block's 8 waves)
  const int h = ht >> 1;
  const int rt = ht & 1;
  const int mt = wave & 1023;
  const int row0 = mt * 16;
  const int r0 = rt * 64;
  const float coef = 0.08838834764831845f;            // 1/sqrt(128)

  v8f c[4];
#pragma unroll
  for (int n = 0; n < 4; ++n) {
    float bv = br[h * NRFF + r0 + n * 16 + (lane & 15)];
#pragma unroll
    for (int r = 0; r < 8; ++r) c[n][r] = bv;         // bias pre-loaded into C
  }

#pragma unroll
  for (int k0 = 0; k0 < NDK; k0 += 32) {
    v16bf a = load_frag(Q + h * NDK, NDIM, row0, k0, lane);
#pragma unroll
    for (int n = 0; n < 4; ++n) {
      v16bf b = load_frag(WrT + (size_t)h * NRFF * NDK, NDK, r0 + n * 16, k0, lane);
      c[n] = wmma_bf16(a, b, c[n]);
    }
  }

  const int mhi = 8 * (lane >> 4);
  const int ncol = lane & 15;
#pragma unroll
  for (int n = 0; n < 4; ++n)
#pragma unroll
    for (int r = 0; r < 8; ++r) {
      float p = c[n][r];
      size_t row = (size_t)(row0 + mhi + r);
      int rr = r0 + n * 16 + ncol;
      size_t idx = row * (size_t)(NHEADS * NFEAT) + h * NFEAT + rr;
      feat[idx]        = (__bf16)(cosf(p) * coef);
      feat[idx + NRFF] = (__bf16)(sinf(p) * coef);
    }
}

// ---- Kv partial sums: KvPart[bh][chunk][d][f] = sum_{n in chunk} kf[n,f]*v[n,d]
// grid = NBH*KVCHUNK blocks, 256 threads (thread = f), v tile staged in LDS.
__global__ void kv_partial(const __bf16* __restrict__ kf, const __bf16* __restrict__ v,
                           float* __restrict__ KvPart, float* __restrict__ ksumPart) {
  const int blk = blockIdx.x;
  const int bh = blk / KVCHUNK;
  const int chunk = blk % KVCHUNK;
  const int b = bh >> 4;
  const int h = bh & 15;
  const int f = threadIdx.x;
  const int nbase = b * NSEQ + chunk * KVSPAN;

  __shared__ float vld[32 * NDK];
  float acc[NDK];
#pragma unroll
  for (int d = 0; d < NDK; ++d) acc[d] = 0.f;
  float s = 0.f;

  for (int n0 = 0; n0 < KVSPAN; n0 += 32) {
    for (int i = threadIdx.x; i < 32 * NDK; i += 256) {
      int nn = i >> 6, dd = i & 63;
      vld[i] = (float)v[(size_t)(nbase + n0 + nn) * NDIM + h * NDK + dd];
    }
    __syncthreads();
    for (int nn = 0; nn < 32; ++nn) {
      float kv = (float)kf[(size_t)(nbase + n0 + nn) * (NHEADS * NFEAT) + h * NFEAT + f];
      s += kv;
#pragma unroll
      for (int d = 0; d < NDK; ++d) acc[d] += kv * vld[nn * NDK + d];
    }
    __syncthreads();
  }
#pragma unroll
  for (int d = 0; d < NDK; ++d)
    KvPart[((size_t)blk * NDK + d) * NFEAT + f] = acc[d];
  ksumPart[(size_t)blk * NFEAT + f] = s;
}

// ---- reduce partials -> KvT bf16 [bh][d][f] and ksumB bf16 [bh][16][f] -----
// ksumB row 0 = ksum, rows 1..15 zero (it is consumed as a WMMA B fragment).
__global__ void kv_reduce(const float* __restrict__ KvPart, const float* __restrict__ ksumPart,
                          __bf16* __restrict__ KvT, __bf16* __restrict__ ksumB) {
  int idx = blockIdx.x * blockDim.x + threadIdx.x;    // NBH*NDK*NFEAT = 524288
  if (idx >= NBH * NDK * NFEAT) return;
  int bh = idx >> 14;                                 // /(64*256)
  int rem = idx & 16383;
  float a = 0.f;
  for (int cc = 0; cc < KVCHUNK; ++cc)
    a += KvPart[(size_t)(bh * KVCHUNK + cc) * (NDK * NFEAT) + rem];
  KvT[idx] = (__bf16)a;
  if (rem < NFEAT) {                                  // d == 0 threads own ksum[f]
    float s = 0.f;
    for (int cc = 0; cc < KVCHUNK; ++cc)
      s += ksumPart[(size_t)(bh * KVCHUNK + cc) * NFEAT + rem];
    __bf16* kb = ksumB + (size_t)bh * 16 * NFEAT + rem;
    kb[0] = (__bf16)s;
#pragma unroll
    for (int r = 1; r < 16; ++r) kb[(size_t)r * NFEAT] = (__bf16)0.f;
  }
}

// ---- attention: attn[row][h*64+d] = (qf @ Kv) / (qf . ksum + 1e-6) ---------
// WMMA K=256; a 5th accumulator (B = ksumB) computes the denominator, which is
// broadcast across the tile with __shfl (ds_bpermute). Waves in a block share h.
__global__ void attn_kernel(const __bf16* __restrict__ qf, const __bf16* __restrict__ KvT,
                            const __bf16* __restrict__ ksumB, __bf16* __restrict__ attn) {
  const int lane = threadIdx.x & 31;
  const int wave = blockIdx.x * (blockDim.x >> 5) + (threadIdx.x >> 5);
  const int h = wave >> 10;         // same for the block's 8 waves
  const int mt = wave & 1023;
  const int row0 = mt * 16;
  const int b = row0 >> 13;
  const int bh = b * NHEADS + h;
  const __bf16* Bb = KvT + (size_t)bh * NDK * NFEAT;
  const __bf16* Bz = ksumB + (size_t)bh * 16 * NFEAT;

  v8f c[4], cz;
#pragma unroll
  for (int n = 0; n < 4; ++n)
#pragma unroll
    for (int r = 0; r < 8; ++r) c[n][r] = 0.f;
#pragma unroll
  for (int r = 0; r < 8; ++r) cz[r] = 0.f;

#pragma unroll
  for (int k0 = 0; k0 < NFEAT; k0 += 32) {
    v16bf a = load_frag(qf + h * NFEAT, NHEADS * NFEAT, row0, k0, lane);
#pragma unroll
    for (int n = 0; n < 4; ++n) {
      v16bf bb = load_frag(Bb, NFEAT, n * 16, k0, lane);
      c[n] = wmma_bf16(a, bb, c[n]);
    }
    v16bf bz = load_frag(Bz, NFEAT, 0, k0, lane);
    cz = wmma_bf16(a, bz, cz);
  }

  const int mhi = 8 * (lane >> 4);
  const int ncol = lane & 15;
  // denominator lives in column 0 -> lanes 0 (rows 0-7) and 16 (rows 8-15)
  float zv[8];
#pragma unroll
  for (int r = 0; r < 8; ++r) {
    float den = __shfl(cz[r], lane & 16, 32);
    zv[r] = 1.f / (den + 1e-6f);
  }
#pragma unroll
  for (int n = 0; n < 4; ++n)
#pragma unroll
    for (int r = 0; r < 8; ++r)
      attn[(size_t)(row0 + mhi + r) * NDIM + h * NDK + n * 16 + ncol] =
          (__bf16)(c[n][r] * zv[r]);
}

// ---------------------------------------------------------------------------
extern "C" void kernel_launch(void* const* d_in, const int* in_sizes, int n_in,
                              void* d_out, int out_size, void* d_ws, size_t ws_size,
                              hipStream_t stream) {
  const float* x  = (const float*)d_in[0];
  const float* Wq = (const float*)d_in[1];
  const float* Wk = (const float*)d_in[2];
  const float* Wv = (const float*)d_in[3];
  const float* Wp = (const float*)d_in[4];
  const float* bp = (const float*)d_in[5];
  const float* Wr = (const float*)d_in[6];
  const float* br = (const float*)d_in[7];
  float* out = (float*)d_out;

  char* ws = (char*)d_ws;
  size_t off = 0;
  auto carve = [&](size_t bytes) {
    char* p = ws + off;
    off += (bytes + 255) & ~(size_t)255;
    return p;
  };
  __bf16* xb    = (__bf16*)carve((size_t)NROWS * NDIM * 2);
  __bf16* Wqb   = (__bf16*)carve((size_t)NDIM * NDIM * 2);
  __bf16* Wkb   = (__bf16*)carve((size_t)NDIM * NDIM * 2);
  __bf16* Wvb   = (__bf16*)carve((size_t)NDIM * NDIM * 2);
  __bf16* Wpb   = (__bf16*)carve((size_t)NDIM * NDIM * 2);
  __bf16* WrTb  = (__bf16*)carve((size_t)NHEADS * NRFF * NDK * 2);
  __bf16* qb    = (__bf16*)carve((size_t)NROWS * NDIM * 2);
  __bf16* kb    = (__bf16*)carve((size_t)NROWS * NDIM * 2);
  __bf16* vb    = (__bf16*)carve((size_t)NROWS * NDIM * 2);
  __bf16* qfb   = (__bf16*)carve((size_t)NROWS * NHEADS * NFEAT * 2);
  __bf16* kfb   = (__bf16*)carve((size_t)NROWS * NHEADS * NFEAT * 2);
  float*  KvPart  = (float*)carve((size_t)NBH * KVCHUNK * NDK * NFEAT * 4);
  float*  ksumPart= (float*)carve((size_t)NBH * KVCHUNK * NFEAT * 4);
  __bf16* KvTb  = (__bf16*)carve((size_t)NBH * NDK * NFEAT * 2);
  __bf16* ksumB = (__bf16*)carve((size_t)NBH * 16 * NFEAT * 2);
  __bf16* attn  = (__bf16*)carve((size_t)NROWS * NDIM * 2);
  (void)ws_size; (void)in_sizes; (void)n_in; (void)out_size;

  // 1) precision conversions (vectorized, element counts are multiples of 4)
  cvt_f32_bf16<<<4096, 256, 0, stream>>>(x, xb, NROWS * NDIM / 4);
  cvt_f32_bf16<<<1024, 256, 0, stream>>>(Wq, Wqb, NDIM * NDIM / 4);
  cvt_f32_bf16<<<1024, 256, 0, stream>>>(Wk, Wkb, NDIM * NDIM / 4);
  cvt_f32_bf16<<<1024, 256, 0, stream>>>(Wv, Wvb, NDIM * NDIM / 4);
  cvt_f32_bf16<<<1024, 256, 0, stream>>>(Wp, Wpb, NDIM * NDIM / 4);
  wr_transpose<<<(NHEADS * NRFF * NDK + 255) / 256, 256, 0, stream>>>(Wr, WrTb);

  // 2) QKV projections: 512 m-tiles x 16 n-strips = 8192 waves -> 1024 blocks
  gemm_xwT<<<1024, 256, 0, stream>>>(xb, Wqb, qb, nullptr, nullptr);
  gemm_xwT<<<1024, 256, 0, stream>>>(xb, Wkb, kb, nullptr, nullptr);
  gemm_xwT<<<1024, 256, 0, stream>>>(xb, Wvb, vb, nullptr, nullptr);

  // 3) RFF features: 1024 m-tiles x 32 (h,rt) = 32768 waves -> 4096 blocks
  rff_kernel<<<4096, 256, 0, stream>>>(qb, WrTb, br, qfb);
  rff_kernel<<<4096, 256, 0, stream>>>(kb, WrTb, br, kfb);

  // 4) Kv/ksum: partial sums over 32 sequence chunks, then parallel reduce
  kv_partial<<<NBH * KVCHUNK, 256, 0, stream>>>(kfb, vb, KvPart, ksumPart);
  kv_reduce<<<(NBH * NDK * NFEAT + 255) / 256, 256, 0, stream>>>(KvPart, ksumPart, KvTb, ksumB);

  // 5) attention output + fused normalizer: 16384 waves -> 2048 blocks
  attn_kernel<<<2048, 256, 0, stream>>>(qfb, KvTb, ksumB, attn);

  // 6) output projection with bias, f32 result
  gemm_xwT<<<1024, 256, 0, stream>>>(attn, Wpb, nullptr, out, bp);
}
